// CrossAttLayer_13675175870586
// MI455X (gfx1250) — matrix operand order repaired and checked by reading, not compiled, not run on hardware
//
#include <hip/hip_runtime.h>
#include <hip/hip_bf16.h>
#include <stdint.h>

// ---------------------------------------------------------------------------
// CDNA5 (gfx1250) wave32 WMMA bf16 implementation of the cross-attention layer.
// fp32 operands converted to bf16 with native HW converts (v_cvt_pk_bf16_f32),
// f32 accumulation via v_wmma_f32_16x16x32_bf16. Attention uses an online
// (flash) softmax with branchless masking.
// ---------------------------------------------------------------------------

typedef __attribute__((ext_vector_type(16))) __bf16 v16bf;
typedef __attribute__((ext_vector_type(8)))  float  v8f;

// Hardware bf16 convert (RNE) — lowers to v_cvt_*bf16_f32 on gfx1250.
__device__ __forceinline__ unsigned short f2bf(float f) {
  union { __bf16 h; unsigned short u; } c;
  c.h = (__bf16)f;
  return c.u;
}

__device__ __forceinline__ v8f wmma_bf16(v16bf a, v16bf b, v8f c) {
  return __builtin_amdgcn_wmma_f32_16x16x32_bf16(
      /*neg_a=*/false, a, /*neg_b=*/false, b,
      /*c_mod=*/(short)0, c, /*reuse_a=*/false, /*reuse_b=*/false);
}

// A-fragment (16x32) from f32 row-major source (row stride ld floats).
// ISA layout: lanes 0-15 -> row=lane, K {k0..k0+7, k0+16..k0+23};
//             lanes 16-31 -> row=lane-16, K shifted by +8.
__device__ __forceinline__ v16bf load_afrag_f32(const float* base, int ld, int k0) {
  int lane = threadIdx.x & 31;
  const float* p = base + (size_t)(lane & 15) * ld + k0 + ((lane >> 4) << 3);
  const float4* q = (const float4*)p;
  float4 a0 = q[0], a1 = q[1], b0 = q[4], b1 = q[5];
  v16bf v;
  v[0] = (__bf16)a0.x; v[1] = (__bf16)a0.y; v[2] = (__bf16)a0.z; v[3] = (__bf16)a0.w;
  v[4] = (__bf16)a1.x; v[5] = (__bf16)a1.y; v[6] = (__bf16)a1.z; v[7] = (__bf16)a1.w;
  v[8] = (__bf16)b0.x; v[9] = (__bf16)b0.y; v[10] = (__bf16)b0.z; v[11] = (__bf16)b0.w;
  v[12] = (__bf16)b1.x; v[13] = (__bf16)b1.y; v[14] = (__bf16)b1.z; v[15] = (__bf16)b1.w;
  return v;
}

// A-fragment (16x32) from bf16 row-major source.
__device__ __forceinline__ v16bf load_afrag_bf16(const unsigned short* base, int ld, int k0) {
  int lane = threadIdx.x & 31;
  const unsigned short* p = base + (size_t)(lane & 15) * ld + k0 + ((lane >> 4) << 3);
  union { uint4 q[2]; v16bf v; } u;
  u.q[0] = *(const uint4*)p;        // k0..k0+7
  u.q[1] = *(const uint4*)(p + 16); // k0+16..k0+23
  return u.v;
}

// B-fragment (32x16): element (k,n) sourced from a (N,K) row-major array so
// each lane reads K contiguously. Layout (from ISA B tables): lane = N column,
// lanes 0-15 hold K {k0..k0+7, k0+16..k0+23}, lanes 16-31 shifted by +8.
__device__ __forceinline__ v16bf load_bfrag_bf16(const unsigned short* base, int ld, int k0) {
  int lane = threadIdx.x & 31;
  const unsigned short* p = base + (size_t)(lane & 15) * ld + k0 + ((lane >> 4) << 3);
  union { uint4 q[2]; v16bf v; } u;
  u.q[0] = *(const uint4*)p;
  u.q[1] = *(const uint4*)(p + 16);
  return u.v;
}

// ---------------------------------------------------------------------------
// Weight transpose + bf16: W (K,N) f32 -> WT (N,K) bf16
// ---------------------------------------------------------------------------
__global__ void wtrans_kernel(const float* __restrict__ W, unsigned short* __restrict__ WT,
                              int K, int N) {
  int idx = blockIdx.x * 256 + threadIdx.x;
  if (idx >= K * N) return;
  int n = idx / K, k = idx - n * K;
  WT[(size_t)n * K + k] = f2bf(W[(size_t)k * N + n]);
}

// ---------------------------------------------------------------------------
// Generic GEMM: out = epilogue(relu(X @ W + bias))
//  MODE 0: plain (M,N) f32 out, optional residual (HAS_RES) added after relu
//  MODE 1: K-projection -> f32 (T,H,R,Dh) + bf16 same layout
//  MODE 2: V-projection -> f32 (T,H,R,Dh) + bf16 transposed (T,H,Dh,R)
//  MODE 3: Q-projection -> relu * q_scale -> f32 (T,H,A,Dh) + bf16 same layout
// X rows are (NR, T) flattened: m = r*TDIM + t.
// ---------------------------------------------------------------------------
template <int KDIM, int MODE, bool HAS_RES>
__global__ void gemm_kernel(const float* __restrict__ X, const unsigned short* __restrict__ WT,
                            const float* __restrict__ bias, int M, int N, int NR,
                            float* __restrict__ out_plain, const float* __restrict__ resid,
                            float* __restrict__ out_f32, unsigned short* __restrict__ out_bf,
                            const float* __restrict__ qscale) {
  constexpr int TDIM = 91;
  int wave = threadIdx.x >> 5, lane = threadIdx.x & 31;
  int m0 = blockIdx.x * 128 + wave * 16;
  int n0 = blockIdx.y * 16;
  if (m0 >= M) return;
  const float* Xb = X + (size_t)m0 * KDIM;
  const unsigned short* Wb = WT + (size_t)n0 * KDIM;

  v8f acc = {0.f, 0.f, 0.f, 0.f, 0.f, 0.f, 0.f, 0.f};
#pragma unroll
  for (int k0 = 0; k0 < KDIM; k0 += 32) {
    v16bf a = load_afrag_f32(Xb, KDIM, k0);
    v16bf b = load_bfrag_bf16(Wb, KDIM, k0);
    acc = wmma_bf16(a, b, acc);
  }

  int col = lane & 15;
  int n = n0 + col;
  int rh = (lane >> 4) << 3; // +8 row offset for upper half-wave
  float bn = bias[n];

  if (MODE == 0) {
#pragma unroll
    for (int i = 0; i < 8; ++i) {
      int m = m0 + i + rh;
      float v = fmaxf(acc[i] + bn, 0.f);
      if (HAS_RES) v += resid[(size_t)m * N + n];
      out_plain[(size_t)m * N + n] = v;
    }
  } else {
    int h = n >> 6, d = n & 63;
    float qs = (MODE == 3) ? qscale[d] : 1.f;
#pragma unroll
    for (int i = 0; i < 8; ++i) {
      int m = m0 + i + rh;
      int r = m / TDIM, t = m - r * TDIM; // TDIM is constexpr -> mul/shift
      float v = fmaxf(acc[i] + bn, 0.f) * qs;
      size_t o = (((size_t)t * 4 + h) * NR + r) * 64 + d;
      out_f32[o] = v;
      unsigned short bf = f2bf(v);
      if (MODE == 2)
        out_bf[(((size_t)t * 4 + h) * 64 + d) * (size_t)NR + r] = bf; // (T,H,Dh,R)
      else
        out_bf[o] = bf;
    }
  }
}

// ---------------------------------------------------------------------------
// Attention: per (t,h) block; blockIdx.y selects agent half (128 rows).
// Each wave owns a 16-row Q tile, streams R in 32-wide chunks with online
// softmax, accumulates O (16x64) in f32 WMMA accumulators. Masking is fully
// branchless. __launch_bounds__(256,1): this kernel is WMMA-latency bound,
// so trade occupancy for a spill-free register allocation (Q fragments and
// softmax state must stay resident across 32 chunk iterations).
// ---------------------------------------------------------------------------
__global__ void __launch_bounds__(256, 1)
attn_kernel(const unsigned short* __restrict__ Qbf,  // (T,H,A,Dh)
            const unsigned short* __restrict__ Kbf,  // (T,H,R,Dh)
            const unsigned short* __restrict__ Vbt,  // (T,H,Dh,R)
            const unsigned char* __restrict__ am,    // (A,R)
            const unsigned char* __restrict__ pm,    // (A,T)
            const unsigned char* __restrict__ rv,    // (R,T)
            float* __restrict__ Y1) {                // (A,T,F)
  constexpr int T = 91, A = 256, R = 1024;
  int th = blockIdx.x;
  int t = th >> 2, h = th & 3;
  int wave = threadIdx.x >> 5, lane = threadIdx.x & 31;
  int a0 = blockIdx.y * 128 + wave * 16;
  int col = lane & 15;
  int rh = (lane >> 4) << 3;

  // Per-wave 16x32 bf16 P staging; strictly private per wave, so no block
  // barriers are required (per-wave LDS ordering is in-order in HW).
  __shared__ unsigned short Pl[8 * 512];
  unsigned short* pw = &Pl[wave * 512];

  const unsigned short* Qb = Qbf + (((size_t)t * 4 + h) * A + a0) * 64;
  const unsigned short* Kb = Kbf + ((size_t)t * 4 + h) * R * 64;
  const unsigned short* Vb = Vbt + ((size_t)t * 4 + h) * 64 * R;

  v16bf qa0 = load_afrag_bf16(Qb, 64, 0);
  v16bf qa1 = load_afrag_bf16(Qb, 64, 32);

  // pack the 8 per-row padding-mask bytes into one bitmask word
  unsigned int pmbits = 0;
#pragma unroll
  for (int i = 0; i < 8; ++i)
    pmbits |= (pm[(size_t)(a0 + i + rh) * T + t] ? 1u : 0u) << i;

  float mrow[8], lrow[8];
  v8f O[4];
#pragma unroll
  for (int i = 0; i < 8; ++i) { mrow[i] = -3.0e38f; lrow[i] = 0.f; }
#pragma unroll
  for (int dt = 0; dt < 4; ++dt)
#pragma unroll
    for (int i = 0; i < 8; ++i) O[dt][i] = 0.f;

  for (int r0 = 0; r0 < R; r0 += 32) {
    // prefetch next chunk's K rows and V rows (global_prefetch_b8)
    if (r0 + 32 < R) {
      __builtin_prefetch(Kb + (size_t)(r0 + 32) * 64 + col * 4, 0, 3);
      __builtin_prefetch(Vb + (size_t)col * R + r0 + 32, 0, 3);
    }

    // energy tiles: S = Q @ K^T / sqrt(H), two 16x16 column tiles
    v8f s0 = {0.f, 0.f, 0.f, 0.f, 0.f, 0.f, 0.f, 0.f};
    v8f s1 = {0.f, 0.f, 0.f, 0.f, 0.f, 0.f, 0.f, 0.f};
    {
      v16bf kb;
      kb = load_bfrag_bf16(Kb + (size_t)r0 * 64, 64, 0);          s0 = wmma_bf16(qa0, kb, s0);
      kb = load_bfrag_bf16(Kb + (size_t)r0 * 64, 64, 32);         s0 = wmma_bf16(qa1, kb, s0);
      kb = load_bfrag_bf16(Kb + (size_t)(r0 + 16) * 64, 64, 0);   s1 = wmma_bf16(qa0, kb, s1);
      kb = load_bfrag_bf16(Kb + (size_t)(r0 + 16) * 64, 64, 32);  s1 = wmma_bf16(qa1, kb, s1);
    }

    // branchless masks: unconditional byte loads, bitwise combine
    unsigned int rv0 = rv[(size_t)(r0 + col) * T + t];
    unsigned int rv1 = rv[(size_t)(r0 + 16 + col) * T + t];
    unsigned int am0[8], am1[8];
#pragma unroll
    for (int i = 0; i < 8; ++i) {
      size_t ab = (size_t)(a0 + i + rh) * R + r0 + col;
      am0[i] = am[ab];
      am1[i] = am[ab + 16];
    }

    float e0[8], e1[8], mx[8];
#pragma unroll
    for (int i = 0; i < 8; ++i) {
      unsigned int pb = (pmbits >> i) & 1u;
      unsigned int v0 = pb & rv0 & am0[i];
      unsigned int v1 = pb & rv1 & am1[i];
      e0[i] = v0 ? s0[i] * 0.5f : -1e10f;   // scale = 1/sqrt(H) = 0.5
      e1[i] = v1 ? s1[i] * 0.5f : -1e10f;
      mx[i] = fmaxf(e0[i], e1[i]);
    }
    // row-max across the 16 lanes holding the same rows
#pragma unroll
    for (int i = 0; i < 8; ++i) {
      mx[i] = fmaxf(mx[i], __shfl_xor(mx[i], 1, 32));
      mx[i] = fmaxf(mx[i], __shfl_xor(mx[i], 2, 32));
      mx[i] = fmaxf(mx[i], __shfl_xor(mx[i], 4, 32));
      mx[i] = fmaxf(mx[i], __shfl_xor(mx[i], 8, 32));
    }
    float al[8], ps[8];
#pragma unroll
    for (int i = 0; i < 8; ++i) {
      float mn = fmaxf(mrow[i], mx[i]);
      al[i] = __expf(mrow[i] - mn);
      mrow[i] = mn;
      e0[i] = __expf(e0[i] - mn);
      e1[i] = __expf(e1[i] - mn);
      ps[i] = e0[i] + e1[i];
    }
#pragma unroll
    for (int i = 0; i < 8; ++i) {
      ps[i] += __shfl_xor(ps[i], 1, 32);
      ps[i] += __shfl_xor(ps[i], 2, 32);
      ps[i] += __shfl_xor(ps[i], 4, 32);
      ps[i] += __shfl_xor(ps[i], 8, 32);
      lrow[i] = lrow[i] * al[i] + ps[i];
    }
#pragma unroll
    for (int dt = 0; dt < 4; ++dt)
#pragma unroll
      for (int i = 0; i < 8; ++i) O[dt][i] *= al[i];

    // stage P (16x32) into LDS to re-layout D-tile -> A-fragment
#pragma unroll
    for (int i = 0; i < 8; ++i) {
      int r = i + rh;
      pw[r * 32 + col] = f2bf(e0[i]);
      pw[r * 32 + 16 + col] = f2bf(e1[i]);
    }
    v16bf pa;
    {
      const unsigned short* pp = pw + (size_t)(lane & 15) * 32 + ((lane >> 4) << 3);
      union { uint4 q[2]; v16bf v; } u;
      u.q[0] = *(const uint4*)pp;
      u.q[1] = *(const uint4*)(pp + 16);
      pa = u.v;
    }
    // O += P @ V_chunk : 4 Dh tiles
#pragma unroll
    for (int dt = 0; dt < 4; ++dt) {
      v16bf vb = load_bfrag_bf16(Vb + (size_t)dt * 16 * R, R, r0);
      O[dt] = wmma_bf16(pa, vb, O[dt]);
    }
  }

  // normalize and write Y1 (A,T,F)
#pragma unroll
  for (int i = 0; i < 8; ++i) {
    float inv = 1.f / lrow[i];
    int a = a0 + i + rh;
#pragma unroll
    for (int dt = 0; dt < 4; ++dt) {
      Y1[((size_t)a * T + t) * 256 + h * 64 + dt * 16 + col] = O[dt][i] * inv;
    }
  }
}

// ---------------------------------------------------------------------------
// LayerNorm over last dim (256) -> Z
// ---------------------------------------------------------------------------
__global__ void ln_kernel(const float* __restrict__ F2, const float* __restrict__ g,
                          const float* __restrict__ b, float* __restrict__ Z) {
  int row = blockIdx.x;
  int tid = threadIdx.x;
  __shared__ float red[256];
  float x = F2[(size_t)row * 256 + tid];
  red[tid] = x;
  __syncthreads();
  for (int s = 128; s > 0; s >>= 1) { if (tid < s) red[tid] += red[tid + s]; __syncthreads(); }
  float mu = red[0] * (1.f / 256.f);
  __syncthreads();
  float d = x - mu;
  red[tid] = d * d;
  __syncthreads();
  for (int s = 128; s > 0; s >>= 1) { if (tid < s) red[tid] += red[tid + s]; __syncthreads(); }
  float var = red[0] * (1.f / 256.f);
  Z[(size_t)row * 256 + tid] = d * rsqrtf(var + 1e-5f) * g[tid] + b[tid];
}

// ---------------------------------------------------------------------------
extern "C" void kernel_launch(void* const* d_in, const int* in_sizes, int n_in,
                              void* d_out, int out_size, void* d_ws, size_t ws_size,
                              hipStream_t stream) {
  (void)in_sizes; (void)n_in; (void)out_size; (void)ws_size;
  const int T = 91, F = 256, A = 256, R = 1024;
  const int MT_rg = R * T;    // 93184
  const int MT_ag = A * T;    // 23296

  const float* agent = (const float*)d_in[0];
  const float* rg    = (const float*)d_in[1];
  const unsigned char* am  = (const unsigned char*)d_in[2];
  const unsigned char* pmk = (const unsigned char*)d_in[3];
  const unsigned char* rvm = (const unsigned char*)d_in[4];
  const float* Wk = (const float*)d_in[5];  const float* bk = (const float*)d_in[6];
  const float* Wv = (const float*)d_in[7];  const float* bv = (const float*)d_in[8];
  const float* Wq = (const float*)d_in[9];  const float* bq = (const float*)d_in[10];
  const float* qscale = (const float*)d_in[11];
  const float* Wy = (const float*)d_in[12]; const float* by = (const float*)d_in[13];
  const float* Wf1 = (const float*)d_in[14]; const float* bf1 = (const float*)d_in[15];
  const float* Wf2 = (const float*)d_in[16]; const float* bf2 = (const float*)d_in[17];
  const float* lng = (const float*)d_in[18]; const float* lnb = (const float*)d_in[19];

  // workspace carve-up
  char* ws = (char*)d_ws;
  size_t off = 0;
  auto alloc = [&](size_t bytes) -> void* {
    void* p = ws + off;
    off = (off + bytes + 255) & ~(size_t)255;
    return p;
  };
  unsigned short* WkT  = (unsigned short*)alloc((size_t)F * F * 2);
  unsigned short* WvT  = (unsigned short*)alloc((size_t)F * F * 2);
  unsigned short* WqT  = (unsigned short*)alloc((size_t)F * F * 2);
  unsigned short* WyT  = (unsigned short*)alloc((size_t)F * F * 2);
  unsigned short* Wf1T = (unsigned short*)alloc((size_t)F * 1024 * 2);
  unsigned short* Wf2T = (unsigned short*)alloc((size_t)1024 * F * 2);
  unsigned short* Qbf  = (unsigned short*)alloc((size_t)T * 4 * A * 64 * 2);
  unsigned short* Kbf  = (unsigned short*)alloc((size_t)T * 4 * R * 64 * 2);
  unsigned short* Vbt  = (unsigned short*)alloc((size_t)T * 4 * 64 * R * 2);
  float* Y1   = (float*)alloc((size_t)MT_ag * F * 4);
  float* Sbuf = (float*)alloc((size_t)MT_ag * F * 4);
  float* F1   = (float*)alloc((size_t)MT_ag * 1024 * 4);
  float* F2   = Y1; // Y1 is dead after the Wy GEMM; reuse its buffer

  // output carve-up: (Z, Qp, Kp, Vp)
  float* outZ = (float*)d_out;
  float* Qp = outZ + (size_t)A * T * F;
  float* Kp = Qp + (size_t)T * 4 * A * 64;
  float* Vp = Kp + (size_t)T * 4 * R * 64;

  dim3 blk(256);

  // 1) weight transposes to bf16 (N,K)
  wtrans_kernel<<<(F * F + 255) / 256, blk, 0, stream>>>(Wk, WkT, F, F);
  wtrans_kernel<<<(F * F + 255) / 256, blk, 0, stream>>>(Wv, WvT, F, F);
  wtrans_kernel<<<(F * F + 255) / 256, blk, 0, stream>>>(Wq, WqT, F, F);
  wtrans_kernel<<<(F * F + 255) / 256, blk, 0, stream>>>(Wy, WyT, F, F);
  wtrans_kernel<<<(F * 1024 + 255) / 256, blk, 0, stream>>>(Wf1, Wf1T, F, 1024);
  wtrans_kernel<<<(1024 * F + 255) / 256, blk, 0, stream>>>(Wf2, Wf2T, 1024, F);

  // 2) projections (write Kp/Vp/Qp outputs directly + bf16 copies for attention)
  gemm_kernel<256, 1, false><<<dim3(MT_rg / 128, F / 16), blk, 0, stream>>>(
      rg, WkT, bk, MT_rg, F, R, nullptr, nullptr, Kp, Kbf, nullptr);
  gemm_kernel<256, 2, false><<<dim3(MT_rg / 128, F / 16), blk, 0, stream>>>(
      rg, WvT, bv, MT_rg, F, R, nullptr, nullptr, Vp, Vbt, nullptr);
  gemm_kernel<256, 3, false><<<dim3(MT_ag / 128, F / 16), blk, 0, stream>>>(
      agent, WqT, bq, MT_ag, F, A, nullptr, nullptr, Qp, Qbf, qscale);

  // 3) attention -> Y1 (A,T,F)
  attn_kernel<<<dim3(T * 4, 2), blk, 0, stream>>>(Qbf, Kbf, Vbt, am, pmk, rvm, Y1);

  // 4) S = relu(Y1 @ Wy + by) + agent
  gemm_kernel<256, 0, true><<<dim3(MT_ag / 128, F / 16), blk, 0, stream>>>(
      Y1, WyT, by, MT_ag, F, 0, Sbuf, agent, nullptr, nullptr, nullptr);

  // 5) F1 = relu(S @ Wf1 + bf1)   (N=1024)
  gemm_kernel<256, 0, false><<<dim3(MT_ag / 128, 1024 / 16), blk, 0, stream>>>(
      Sbuf, Wf1T, bf1, MT_ag, 1024, 0, F1, nullptr, nullptr, nullptr, nullptr);

  // 6) F2 = relu(F1 @ Wf2 + bf2)  (K=1024)
  gemm_kernel<1024, 0, false><<<dim3(MT_ag / 128, F / 16), blk, 0, stream>>>(
      F1, Wf2T, bf2, MT_ag, F, 0, F2, nullptr, nullptr, nullptr, nullptr);

  // 7) layernorm -> Z
  ln_kernel<<<MT_ag, blk, 0, stream>>>(F2, lng, lnb, outZ);
}
